// PointnetSAModuleMSG_16947940950606
// MI455X (gfx1250) — matrix operand order repaired and checked by reading, not compiled, not run on hardware
//
#include <hip/hip_runtime.h>

typedef __attribute__((ext_vector_type(16))) _Float16 v16h;
typedef __attribute__((ext_vector_type(8)))  float    v8f;

#define B_      8
#define N_      4096
#define S_      1024
#define C_FEAT  128
#define C_IN    131
#define MID     64
#define C_OUT   256
#define BN_EPSF 1e-5f

// ---------------- workspace layout (bytes, all 256-aligned) ----------------
static constexpr size_t OFF_IDX   = 0;          // 8192*64*4      = 2,097,152
static constexpr size_t OFF_W2H   = 2097152;    // 144*64*2       = 18,432
static constexpr size_t OFF_WCRH  = 2115584;    // 256*160*2      = 81,920
static constexpr size_t OFF_ST1   = 2197504;    // 128*4
static constexpr size_t OFF_AB1   = 2198016;    // 128*4
static constexpr size_t OFF_ST2   = 2198528;    // 262*4 (pad 1280)
static constexpr size_t OFF_AB2   = 2199808;    // 262*4 (pad 1280)
static constexpr size_t OFF_ST3   = 2201088;    // 512*4
static constexpr size_t OFF_AB3   = 2203136;    // 512*4
static constexpr size_t OFF_MAXH  = 2205184;    // 8192*160*2     = 2,621,440
static constexpr size_t OFF_P3    = 4826624;    // 8192*256*4     = 8,388,608
static constexpr size_t OFF_X     = 13215232;   // 524288*131*4   = 274,726,912

// ---------------- FPS: one block per batch, coords LDS-resident ----------------
__global__ __launch_bounds__(1024) void fps_kernel(const float* __restrict__ xyz,
                                                   float* __restrict__ newxyz) {
    __shared__ float xs[N_], ys[N_], zs[N_];
    __shared__ float rv[32];
    __shared__ int   ri[32];
    __shared__ int   lastS;
    const int b = blockIdx.x, t = threadIdx.x;
    for (int j = 0; j < 4; ++j) {
        int p = t + j * 1024;
        const float* s = &xyz[(size_t)(b * N_ + p) * 3];
        xs[p] = s[0]; ys[p] = s[1]; zs[p] = s[2];
    }
    __syncthreads();
    if (t == 0) {
        float* o = &newxyz[(size_t)b * S_ * 3];
        o[0] = xs[0]; o[1] = ys[0]; o[2] = zs[0];
    }
    float d[4] = {1e10f, 1e10f, 1e10f, 1e10f};
    int last = 0;
    const int lane = t & 31, wid = t >> 5;
    for (int it = 1; it < S_; ++it) {
        const float lx = xs[last], ly = ys[last], lz = zs[last];
        float bv = -1.0f; int bi = 0;
        #pragma unroll
        for (int j = 0; j < 4; ++j) {
            int p = t + j * 1024;
            float dx = xs[p] - lx, dy = ys[p] - ly, dz = zs[p] - lz;
            float dd = dx * dx + dy * dy + dz * dz;
            d[j] = fminf(d[j], dd);
            if (d[j] > bv) { bv = d[j]; bi = p; }
        }
        #pragma unroll
        for (int o = 16; o; o >>= 1) {
            float ov = __shfl_xor(bv, o, 32);
            int   oi = __shfl_xor(bi, o, 32);
            if (ov > bv || (ov == bv && oi < bi)) { bv = ov; bi = oi; }
        }
        if (lane == 0) { rv[wid] = bv; ri[wid] = bi; }
        __syncthreads();
        if (t < 32) {
            float v2 = rv[t]; int i2 = ri[t];
            #pragma unroll
            for (int o = 16; o; o >>= 1) {
                float ov = __shfl_xor(v2, o, 32);
                int   oi = __shfl_xor(i2, o, 32);
                if (ov > v2 || (ov == v2 && oi < i2)) { v2 = ov; i2 = oi; }
            }
            if (t == 0) {
                lastS = i2;
                float* o2 = &newxyz[(size_t)(b * S_ + it) * 3];
                o2[0] = xs[i2]; o2[1] = ys[i2]; o2[2] = zs[i2];
            }
        }
        __syncthreads();
        last = lastS;
    }
}

// ---------------- ball query: one wave per query, ordered ballot-scan ----------------
__global__ __launch_bounds__(256) void ballq_kernel(const float* __restrict__ xyz,
                                                    const float* __restrict__ newxyz,
                                                    int* __restrict__ idx, float r2, int K) {
    const int wid = threadIdx.x >> 5, lane = threadIdx.x & 31;
    const int q = blockIdx.x * 8 + wid;       // 0..8191
    const int b = q >> 10;
    const float* qp = &newxyz[(size_t)q * 3];
    const float qx = qp[0], qy = qp[1], qz = qp[2];
    const float* base = &xyz[(size_t)b * N_ * 3];
    int cnt = 0, firstI = 0;
    bool haveFirst = false;
    for (int s0 = 0; s0 < N_; s0 += 32) {
        int p = s0 + lane;
        float dx = base[p * 3 + 0] - qx;
        float dy = base[p * 3 + 1] - qy;
        float dz = base[p * 3 + 2] - qz;
        bool hit = (dx * dx + dy * dy + dz * dz) < r2;
        unsigned mask = (unsigned)__ballot(hit);
        if (!haveFirst && mask) { firstI = s0 + __ffs(mask) - 1; haveFirst = true; }
        if (hit) {
            int slot = cnt + __popc(mask & ((1u << lane) - 1u));
            if (slot < K) idx[(size_t)q * K + slot] = p;
        }
        cnt += __popc(mask);
        if (cnt >= K) break;
    }
    for (int j = cnt + lane; j < K; j += 32) idx[(size_t)q * K + j] = firstI;
}

// ---------------- BN1 stats: P1 = H*W1^T + b1, accumulate per-channel sum/sumsq ----------------
__global__ __launch_bounds__(256) void stats1_kernel(const float* __restrict__ xyz,
                                                     const float* __restrict__ newxyz,
                                                     const int* __restrict__ idx,
                                                     const float* __restrict__ W1,
                                                     const float* __restrict__ b1,
                                                     float* __restrict__ st, int K) {
    __shared__ float W1s[MID * 10], b1s[MID], binS[MID], binQ[MID];
    const int t = threadIdx.x;
    for (int i = t; i < MID * 10; i += 256) W1s[i] = W1[i];
    if (t < MID) { b1s[t] = b1[t]; binS[t] = 0.f; binQ[t] = 0.f; }
    __syncthreads();
    const int e = blockIdx.x * 256 + t;
    const int q = e / K, k = e % K, b = q >> 10;
    const int i = idx[(size_t)q * K + k], i0 = idx[(size_t)q * K];
    const float* g  = &xyz[(size_t)(b * N_ + i) * 3];
    const float* c0 = &xyz[(size_t)(b * N_ + i0) * 3];
    const float* qp = &newxyz[(size_t)q * 3];
    float rx = g[0] - qp[0], ry = g[1] - qp[1], rz = g[2] - qp[2];
    float H[10];
    H[0] = sqrtf(rx * rx + ry * ry + rz * rz + 1e-12f);
    H[1] = c0[0]; H[2] = c0[1]; H[3] = c0[2];
    H[4] = g[0];  H[5] = g[1];  H[6] = g[2];
    H[7] = rx;    H[8] = ry;    H[9] = rz;
    const int lane = t & 31;
    for (int c = 0; c < MID; ++c) {
        float p = b1s[c];
        #pragma unroll
        for (int j = 0; j < 10; ++j) p += H[j] * W1s[c * 10 + j];
        float s = p, sq = p * p;
        #pragma unroll
        for (int o = 16; o; o >>= 1) { s += __shfl_xor(s, o, 32); sq += __shfl_xor(sq, o, 32); }
        if (lane == 0) { atomicAdd(&binS[c], s); atomicAdd(&binQ[c], sq); }
    }
    __syncthreads();
    if (t < MID) { atomicAdd(&st[t], binS[t]); atomicAdd(&st[MID + t], binQ[t]); }
}

// ---------------- fold BN stats into scale/shift: y = a*x + c ----------------
__global__ void finalize_kernel(const float* __restrict__ st, const float* __restrict__ g,
                                const float* __restrict__ bb, float* __restrict__ ab,
                                int C, float invM) {
    int c = threadIdx.x;
    if (c >= C) return;
    float mean = st[c] * invM;
    float var  = st[C + c] * invM - mean * mean;
    float a = g[c] * rsqrtf(var + BN_EPSF);
    ab[c] = a;
    ab[C + c] = bb[c] - mean * a;
}

// ---------------- main per-element kernel: h1 -> WMMA(64->131) -> x = P2*X, store x ----------------
// block = 288 threads (9 waves); each block: 16 rows (one (b,s), 16 consecutive k),
// wave w computes output column tile w (16 cols, 144 padded >= 131).
__global__ __launch_bounds__(288) void main_kernel(const float* __restrict__ xyz,
                                                   const float* __restrict__ newxyz,
                                                   const int* __restrict__ idx,
                                                   const float* __restrict__ features,
                                                   const float* __restrict__ W1,
                                                   const float* __restrict__ b1,
                                                   const float* __restrict__ ab1,
                                                   const _Float16* __restrict__ W2h,
                                                   const float* __restrict__ b2,
                                                   float* __restrict__ xout, int K) {
    __shared__ float    Hs[16][10];
    __shared__ int      pidx[16];
    __shared__ _Float16 Ah[16 * 64];
    __shared__ float    W1s[640], b1s[64], a1s[64], c1s[64];
    const int t = threadIdx.x;
    const int ebase = blockIdx.x * 16;
    const int q = ebase / K, k0 = ebase % K, b = q >> 10;

    for (int i = t; i < 640; i += 288) W1s[i] = W1[i];
    if (t < 64) { b1s[t] = b1[t]; a1s[t] = ab1[t]; c1s[t] = ab1[64 + t]; }
    if (t < 16) {
        int i  = idx[(size_t)q * K + k0 + t]; pidx[t] = i;
        int i0 = idx[(size_t)q * K];
        const float* g  = &xyz[(size_t)(b * N_ + i) * 3];
        const float* c0 = &xyz[(size_t)(b * N_ + i0) * 3];
        const float* qp = &newxyz[(size_t)q * 3];
        float rx = g[0] - qp[0], ry = g[1] - qp[1], rz = g[2] - qp[2];
        Hs[t][0] = sqrtf(rx * rx + ry * ry + rz * rz + 1e-12f);
        Hs[t][1] = c0[0]; Hs[t][2] = c0[1]; Hs[t][3] = c0[2];
        Hs[t][4] = g[0];  Hs[t][5] = g[1];  Hs[t][6] = g[2];
        Hs[t][7] = rx;    Hs[t][8] = ry;    Hs[t][9] = rz;
    }
    __syncthreads();
    // h1 = relu(a1 * (H*W1^T + b1) + c1), staged to LDS as f16 (row-major 16x64)
    for (int cc = t; cc < 1024; cc += 288) {
        int r = cc >> 6, c = cc & 63;
        float p = b1s[c];
        #pragma unroll
        for (int j = 0; j < 10; ++j) p += Hs[r][j] * W1s[c * 10 + j];
        Ah[r * 64 + c] = (_Float16)fmaxf(0.f, a1s[c] * p + c1s[c]);
    }
    __syncthreads();

    const int lane = t & 31, w = t >> 5;       // w = column tile (0..8)
    const int rowl = lane & 15, hi = lane >> 4;
    v8f acc = {};
    #pragma unroll
    for (int kb = 0; kb < 64; kb += 32) {
        v16h a, bv;
        // A layout (16-bit, 16x32): lane<16 -> K 0..15; lane>=16 -> K 16..31
        int abase = rowl * 64 + kb + hi * 16;
        #pragma unroll
        for (int j = 0; j < 8; ++j) { a[j] = Ah[abase + j]; a[j + 8] = Ah[abase + 8 + j]; }
        // B layout (32x16): lane = column, halves = K 0..15 (+16 for lanes>=16)
        bv = *(const v16h*)&W2h[(size_t)(w * 16 + rowl) * 64 + kb + hi * 16];
        acc = __builtin_amdgcn_wmma_f32_16x16x32_f16(false, a, false, bv, (short)0, acc,
                                                     false, false);
    }
    const int o = w * 16 + rowl;
    if (o < C_IN) {
        const float b2v = b2[o];
        // address-space-pure branches (avoid flat loads): LDS rel vs global feature gather
        float xm[8];
        if (o < 3) {
            #pragma unroll
            for (int v = 0; v < 8; ++v) xm[v] = Hs[v + hi * 8][7 + o];
        } else {
            const float* fp = &features[(size_t)(b * C_FEAT + (o - 3)) * N_];
            #pragma unroll
            for (int v = 0; v < 8; ++v) xm[v] = fp[pidx[v + hi * 8]];
        }
        #pragma unroll
        for (int v = 0; v < 8; ++v) {
            int r = v + hi * 8;                 // D layout: row = v + 8*(lane>=16)
            xout[(size_t)(q * K + k0 + r) * C_IN + o] = (acc[v] + b2v) * xm[v];
        }
    }
}

// ---------------- per-column sum/sumsq over stored matrix (rows x C) ----------------
__global__ void colstats_kernel(const float* __restrict__ x, float* __restrict__ st,
                                int C, int rowsPerBlock) {
    const int t = threadIdx.x;
    if (t >= C) return;
    const size_t r0 = (size_t)blockIdx.x * rowsPerBlock;
    float s = 0.f, sq = 0.f;
    for (int r = 0; r < rowsPerBlock; ++r) {
        if (r + 8 < rowsPerBlock) __builtin_prefetch(&x[(r0 + r + 8) * C + t], 0, 0);
        float v = x[(r0 + r) * C + t];
        s += v; sq += v * v;
    }
    atomicAdd(&st[t], s); atomicAdd(&st[C + t], sq);
}

// ---------------- y = relu(bn2(x)), max over K; store f16 padded to 160 ----------------
__global__ __launch_bounds__(160) void maxk_kernel(const float* __restrict__ x,
                                                   const float* __restrict__ ab2,
                                                   _Float16* __restrict__ maxh, int K) {
    const int t = threadIdx.x, bs = blockIdx.x;
    if (t >= C_IN) { maxh[(size_t)bs * 160 + t] = (_Float16)0.f; return; }
    const float a = ab2[t], c = ab2[C_IN + t];
    float m = 0.f;
    const size_t base = (size_t)bs * K * C_IN + t;
    for (int k = 0; k < K; ++k) {
        if (k + 8 < K) __builtin_prefetch(&x[base + (size_t)(k + 8) * C_IN], 0, 0);
        float v = x[base + (size_t)k * C_IN];
        m = fmaxf(m, fmaxf(0.f, a * v + c));
    }
    maxh[(size_t)bs * 160 + t] = (_Float16)m;
}

// ---------------- final matmul (131->256) via WMMA: P3 = maxed*Wcr^T + bcr ----------------
__global__ __launch_bounds__(256) void mm3_kernel(const _Float16* __restrict__ maxh,
                                                  const _Float16* __restrict__ Wcrh,
                                                  const float* __restrict__ bcr,
                                                  float* __restrict__ P3) {
    const int t = threadIdx.x, lane = t & 31, w = t >> 5;
    const int rowbase = blockIdx.x * 16;
    const int rowl = lane & 15, hi = lane >> 4;
    const _Float16* arow = &maxh[(size_t)(rowbase + rowl) * 160];
    // hoist the 5 A-fragments (K = 160) out of the column-tile loop
    v16h af[5];
    #pragma unroll
    for (int kk = 0; kk < 5; ++kk) {
        int abase = kk * 32 + hi * 16;
        #pragma unroll
        for (int j = 0; j < 8; ++j) { af[kk][j] = arow[abase + j]; af[kk][j + 8] = arow[abase + 8 + j]; }
    }
    #pragma unroll
    for (int ct2 = 0; ct2 < 2; ++ct2) {
        const int ct = w * 2 + ct2;            // 0..15 -> 256 cols
        v8f acc = {};
        #pragma unroll
        for (int kk = 0; kk < 5; ++kk) {
            v16h bv = *(const v16h*)&Wcrh[(size_t)(ct * 16 + rowl) * 160 + kk * 32 + hi * 16];
            acc = __builtin_amdgcn_wmma_f32_16x16x32_f16(false, af[kk], false, bv, (short)0,
                                                         acc, false, false);
        }
        const int o = ct * 16 + rowl;
        const float bb = bcr[o];
        #pragma unroll
        for (int v = 0; v < 8; ++v) {
            int r = v + hi * 8;
            P3[(size_t)(rowbase + r) * C_OUT + o] = acc[v] + bb;
        }
    }
}

// ---------------- out[b, 256*scale + c, s] = relu(bn3(P3)) ----------------
__global__ __launch_bounds__(256) void apply3_kernel(const float* __restrict__ P3,
                                                     const float* __restrict__ ab3,
                                                     float* __restrict__ outF, int scOff) {
    const int bi = blockIdx.x;
    const int b = bi >> 8, c = bi & 255;
    const float a = ab3[c], cc = ab3[C_OUT + c];
    float* op = &outF[(size_t)(b * 512 + scOff + c) * S_];
    const float* pp = &P3[(size_t)b * S_ * C_OUT + c];
    for (int s = threadIdx.x; s < S_; s += 256)
        op[s] = fmaxf(0.f, a * pp[(size_t)s * C_OUT] + cc);
}

// ---------------- weight conversion to padded f16 ----------------
__global__ void convert_kernel(const float* __restrict__ W2, const float* __restrict__ Wcr,
                               _Float16* __restrict__ W2h, _Float16* __restrict__ Wcrh) {
    const int t = blockIdx.x * 256 + threadIdx.x;
    if (t < 144 * 64) {                         // W2h: [144][64], rows >=131 zero
        int n = t >> 6, c = t & 63;
        W2h[t] = (_Float16)((n < C_IN) ? W2[n * 64 + c] : 0.f);
    }
    for (int i = t; i < 256 * 160; i += gridDim.x * 256) {  // Wcrh: [256][160], k>=131 zero
        int n = i / 160, k = i % 160;
        Wcrh[i] = (_Float16)((k < C_IN) ? Wcr[n * C_IN + k] : 0.f);
    }
}

__global__ void zero_kernel(float* st1, float* st2, float* st3) {
    const int t = threadIdx.x;
    if (t < 128) st1[t] = 0.f;
    if (t < 262) st2[t] = 0.f;
    if (t < 512) st3[t] = 0.f;
}

extern "C" void kernel_launch(void* const* d_in, const int* in_sizes, int n_in,
                              void* d_out, int out_size, void* d_ws, size_t ws_size,
                              hipStream_t stream) {
    const float* xyz      = (const float*)d_in[0];
    const float* features = (const float*)d_in[1];
    const float* W1       = (const float*)d_in[2];
    const float* b1       = (const float*)d_in[3];
    const float* W2       = (const float*)d_in[4];
    const float* b2       = (const float*)d_in[5];
    const float* Wcr      = (const float*)d_in[6];
    const float* bcr      = (const float*)d_in[7];
    const float* g_map    = (const float*)d_in[8];
    const float* b_map    = (const float*)d_in[9];
    const float* g_rs     = (const float*)d_in[10];
    const float* b_rs     = (const float*)d_in[11];
    const float* g_cr     = (const float*)d_in[12];
    const float* b_cr     = (const float*)d_in[13];

    float* outNew  = (float*)d_out;                 // (8,1024,3)
    float* outFeat = outNew + B_ * S_ * 3;          // (8,512,1024)

    char* ws = (char*)d_ws;
    int*      idx  = (int*)(ws + OFF_IDX);
    _Float16* W2h  = (_Float16*)(ws + OFF_W2H);
    _Float16* Wcrh = (_Float16*)(ws + OFF_WCRH);
    float*    st1  = (float*)(ws + OFF_ST1);
    float*    ab1  = (float*)(ws + OFF_AB1);
    float*    st2  = (float*)(ws + OFF_ST2);
    float*    ab2  = (float*)(ws + OFF_AB2);
    float*    st3  = (float*)(ws + OFF_ST3);
    float*    ab3  = (float*)(ws + OFF_AB3);
    _Float16* maxh = (_Float16*)(ws + OFF_MAXH);
    float*    P3   = (float*)(ws + OFF_P3);
    float*    xbuf = (float*)(ws + OFF_X);

    fps_kernel<<<B_, 1024, 0, stream>>>(xyz, outNew);
    convert_kernel<<<64, 256, 0, stream>>>(W2, Wcr, W2h, Wcrh);

    const float radii[2] = {0.1f, 0.2f};
    const int   Ks[2]    = {32, 64};
    for (int sc = 0; sc < 2; ++sc) {
        const int K = Ks[sc];
        const int M = B_ * S_ * K;               // 262144 / 524288
        zero_kernel<<<1, 512, 0, stream>>>(st1, st2, st3);
        ballq_kernel<<<1024, 256, 0, stream>>>(xyz, outNew, idx, radii[sc] * radii[sc], K);
        stats1_kernel<<<M / 256, 256, 0, stream>>>(xyz, outNew, idx, W1, b1, st1, K);
        finalize_kernel<<<1, 64, 0, stream>>>(st1, g_map + sc * MID, b_map + sc * MID,
                                              ab1, MID, 1.f / (float)M);
        main_kernel<<<M / 16, 288, 0, stream>>>(xyz, outNew, idx, features, W1, b1, ab1,
                                                W2h, b2, xbuf, K);
        colstats_kernel<<<1024, 160, 0, stream>>>(xbuf, st2, C_IN, M / 1024);
        finalize_kernel<<<1, 160, 0, stream>>>(st2, g_rs + sc * C_IN, b_rs + sc * C_IN,
                                               ab2, C_IN, 1.f / (float)M);
        maxk_kernel<<<B_ * S_, 160, 0, stream>>>(xbuf, ab2, maxh, K);
        mm3_kernel<<<(B_ * S_) / 16, 256, 0, stream>>>(maxh, Wcrh, bcr, P3);
        colstats_kernel<<<64, 256, 0, stream>>>(P3, st3, C_OUT, (B_ * S_) / 64);
        finalize_kernel<<<1, 256, 0, stream>>>(st3, g_cr + sc * C_OUT, b_cr + sc * C_OUT,
                                               ab3, C_OUT, 1.f / (float)(B_ * S_));
        apply3_kernel<<<B_ * C_OUT, 256, 0, stream>>>(P3, ab3, outFeat, sc * C_OUT);
    }
}
